// FeatureMapTemporalModule_6339371729606
// MI455X (gfx1250) — matrix operand order repaired and checked by reading, not compiled, not run on hardware
//
#include <hip/hip_runtime.h>
#include <hip/hip_bf16.h>

// ---------------------------------------------------------------------------
// FeatureMapTemporalModule for MI455X (gfx1250, wave32, WMMA)
//
// One workgroup (256 threads = 8 waves) per temporal graph (BN = 784 graphs,
// each T=64 nodes x C=512 features). All three GEMM stages use
// v_wmma_f32_16x16x32_f16 with 4x4 register blocking per wave (each wave owns
// a 64x64 output block; 8 fragment loads feed 16 wmma per K-step).
// LDS ~266KB of the 320KB WGP budget, aliased across stages -> 1 WG/WGP,
// 2 waves/SIMD, so high VGPR budget for the 16 accumulator tiles.
// ---------------------------------------------------------------------------

typedef __attribute__((ext_vector_type(16))) _Float16 v16h;
typedef __attribute__((ext_vector_type(8)))  _Float16 v8h;
typedef __attribute__((ext_vector_type(8)))  float    v8f;

#define T_   64
#define C_   512
#define N_   196
#define NH_  8
#define DH_  64
#define KSEL 19   // int(0.3 * 64)

// A fragment (16x32 f16, MxK): lanes 0-15 row M=lane hold K {0..7,16..23},
// lanes 16-31 row M=lane-16 hold K {8..15,24..31}. Two 16B loads.
__device__ __forceinline__ v16h ldA(const _Float16* base, int stride,
                                    int m0, int k0, int lane) {
  int m = m0 + (lane & 15);
  const _Float16* p = base + m * stride + k0 + ((lane & 16) ? 8 : 0);
  v8h lo = *reinterpret_cast<const v8h*>(p);
  v8h hi = *reinterpret_cast<const v8h*>(p + 16);
  v16h a;
#pragma unroll
  for (int i = 0; i < 8; ++i) { a[i] = lo[i]; a[i + 8] = hi[i]; }
  return a;
}

// B fragment (32x16 f16, KxN): lanes 0-15 column N=lane hold K {0..7,16..23},
// lanes 16-31 hold K {8..15,24..31} (matches ldA). Source is stored
// transposed [N][K]; two 16B loads.
__device__ __forceinline__ v16h ldB(const _Float16* baseT, int stride,
                                    int n0, int k0, int lane) {
  const _Float16* p =
      baseT + (size_t)(n0 + (lane & 15)) * stride + k0 + ((lane & 16) ? 8 : 0);
  v8h lo = *reinterpret_cast<const v8h*>(p);
  v8h hi = *reinterpret_cast<const v8h*>(p + 16);
  v16h b;
#pragma unroll
  for (int i = 0; i < 8; ++i) { b[i] = lo[i]; b[i + 8] = hi[i]; }
  return b;
}

#define WMMA_F16(acc, A, B)                                                    \
  acc = __builtin_amdgcn_wmma_f32_16x16x32_f16(false, (A), false, (B),         \
                                               (short)0, (acc), false, false)

__global__ void fmt_prep(const float* __restrict__ Wg,
                         const float* __restrict__ W2,
                         _Float16* __restrict__ WgT,
                         _Float16* __restrict__ W2T) {
  int idx = blockIdx.x * blockDim.x + threadIdx.x;
  if (idx >= C_ * C_) return;
  int o = idx >> 9;        // output channel (head*64+d for WgT)
  int c = idx & (C_ - 1);  // input channel
  int h = o >> 6, d = o & 63;
  WgT[(size_t)o * C_ + c] = (_Float16)Wg[((size_t)(h * C_ + c)) * DH_ + d];
  W2T[(size_t)o * C_ + c] = (_Float16)W2[(size_t)c * C_ + o];
}

__global__ __launch_bounds__(256, 1)
void fmt_main(const float* __restrict__ x,
              const _Float16* __restrict__ WgT,
              const float* __restrict__ a1, const float* __restrict__ a2,
              const _Float16* __restrict__ W2T,
              const float* __restrict__ b2,
              const float* __restrict__ ln_g, const float* __restrict__ ln_b,
              float* __restrict__ out) {
  extern __shared__ char smem[];
  // [0,64K)    sX16 [64][512] f16   (GEMM1 A)        -- dead after GEMM1
  // [64K,128K) sAtt [8][64][64] f16 (GEMM2 A)        -- dead after GEMM2
  // [0,128K)   sY   [64][512] f32   (aliases the two above, GEMM3 output)
  // [128K,192K) sHT [512][64] f16   (H transposed: GEMM2 B, f1/f2 source)
  // [192K,256K) sTmp [64][512] f16  (GEMM2 out -> GEMM3 A)
  // [256K,..)   small scratch
  _Float16* sX16 = (_Float16*)(smem);
  _Float16* sAtt = (_Float16*)(smem + 65536);
  float*    sY   = (float*)(smem);
  _Float16* sHT  = (_Float16*)(smem + 131072);
  _Float16* sTmp = (_Float16*)(smem + 196608);
  float* mags  = (float*)(smem + 262144);   // [64]
  float* magsP = mags + 64;                 // [64][4]
  int*   sel   = (int*)(magsP + 256);       // [64]
  float* f1    = (float*)(sel + 64);        // [8][64]
  float* f2    = f1 + 512;                  // [8][64]
  float* rsc   = f2 + 512;                  // [512]
  float* rowP  = rsc + 512;                 // [64][4]
  float* rowP2 = rowP + 256;                // [64][4]
  float* rowMu = rowP2 + 256;               // [64]
  float* rowRs = rowMu + 64;                // [64]

  const int bn = blockIdx.x;
  const int b = bn / N_, n = bn % N_;
  const int tid = threadIdx.x;
  const int lane = tid & 31, wave = tid >> 5;

  // ---- Stage 1: gather X (strided) -> f16, per-frame L2 magnitudes --------
  {
    int r = tid >> 2, q = tid & 3;
    const float* xp = x + ((size_t)(b * T_ + r) * C_) * N_ + n;
    float ssq = 0.f;
#pragma unroll 4
    for (int i = 0; i < 128; ++i) {
      int c = q * 128 + i;
      float v = xp[(size_t)c * N_];
      sX16[r * C_ + c] = (_Float16)v;
      ssq += v * v;
    }
    magsP[r * 4 + q] = ssq;
  }
  __syncthreads();
  if (tid < T_) {
    mags[tid] = sqrtf(magsP[tid * 4] + magsP[tid * 4 + 1] +
                      magsP[tid * 4 + 2] + magsP[tid * 4 + 3]);
  }
  __syncthreads();
  // top-k(19) by rank (ties -> lower index, matching lax.top_k)
  if (tid < T_) {
    float mi = mags[tid];
    int cnt = 0;
    for (int j = 0; j < T_; ++j) {
      float mj = mags[j];
      cnt += (mj > mi) || (mj == mi && j < tid);
    }
    sel[tid] = (cnt < KSEL) ? 1 : 0;
  }
  __syncthreads();

  // ---- Stage 2: GEMM1  H = X @ Wcat  (M=64,N=512,K=512), store H^T --------
  // 4x4 register-blocked: per K-step load 4 A + 4 B fragments -> 16 wmma.
  {
    const int col0 = wave * 64;  // each wave owns a 64-wide N strip
    v8f acc[4][4] = {};
    for (int kt = 0; kt < 16; ++kt) {
      if (kt + 2 < 16) {  // prefetch weight stream 2 K-steps ahead (L2->WGP$)
        __builtin_prefetch(WgT + (size_t)(col0 + (lane & 15)) * C_ +
                               (kt + 2) * 32, 0, 0);
      }
      v16h a[4], bb[4];
#pragma unroll
      for (int mt = 0; mt < 4; ++mt)
        a[mt] = ldA(sX16, C_, mt * 16, kt * 32, lane);
#pragma unroll
      for (int nt = 0; nt < 4; ++nt)
        bb[nt] = ldB(WgT, C_, col0 + nt * 16, kt * 32, lane);
#pragma unroll
      for (int mt = 0; mt < 4; ++mt)
#pragma unroll
        for (int nt = 0; nt < 4; ++nt)
          WMMA_F16(acc[mt][nt], a[mt], bb[nt]);
    }
#pragma unroll
    for (int mt = 0; mt < 4; ++mt)
#pragma unroll
      for (int nt = 0; nt < 4; ++nt) {
        int cc = col0 + nt * 16 + (lane & 15);
        int mo = mt * 16 + ((lane & 16) ? 8 : 0);
#pragma unroll
        for (int i = 0; i < 8; ++i)
          sHT[cc * T_ + (mo + i)] = (_Float16)acc[mt][nt][i];
      }
  }
  __syncthreads();

  // ---- Stage 3: f1/f2 per (head, t) ---------------------------------------
  for (int p = tid; p < NH_ * T_; p += 256) {
    int h = p >> 6, t = p & 63;
    const _Float16* hp = sHT + (size_t)(h * DH_) * T_ + t;
    float s1 = 0.f, s2 = 0.f;
#pragma unroll 4
    for (int d = 0; d < DH_; ++d) {
      float hv = (float)hp[d * T_];
      s1 += hv * a1[h * DH_ + d];
      s2 += hv * a2[h * DH_ + d];
    }
    f1[p] = s1; f2[p] = s2;
  }
  __syncthreads();

  // ---- Stage 4: masked leaky-relu scores + softmax -> sAtt (f16) ----------
  // adj0 = exp(-|...|) > 0 everywhere, so the gate is exactly sel[t]|sel[s].
  for (int p = tid; p < NH_ * T_; p += 256) {
    int h = p >> 6, t = p & 63;
    float ft = f1[p];
    int st = sel[t];
    const float* f2h = f2 + (h << 6);
    float m = -3.0e38f;
    for (int s = 0; s < T_; ++s) {
      if (st || sel[s]) {
        float z = ft + f2h[s];
        float e = z > 0.f ? z : 0.2f * z;
        m = fmaxf(m, e);
      }
    }
    _Float16* ap = sAtt + ((size_t)(h * T_) + t) * T_;
    float sum = 0.f;
    for (int s = 0; s < T_; ++s) {
      float pz = 0.f;
      if (st || sel[s]) {
        float z = ft + f2h[s];
        float e = z > 0.f ? z : 0.2f * z;
        pz = __expf(e - m);
      }
      ap[s] = (_Float16)pz;
      sum += pz;
    }
    float rs = 1.f / sum;
    for (int s = 0; s < T_; ++s) ap[s] = (_Float16)((float)ap[s] * rs);
  }
  __syncthreads();

  // ---- Stage 5: per-head GEMM2  out = att @ H, ELU, signed sqrt -----------
  {
    const int h = wave;  // 8 waves == 8 heads
    const _Float16* A  = sAtt + (size_t)h * T_ * T_;
    const _Float16* Bt = sHT + (size_t)(h * DH_) * T_;  // [d][s], contiguous K
    v8f acc[4][4] = {};
#pragma unroll
    for (int kt = 0; kt < 2; ++kt) {
      v16h a[4], bb[4];
#pragma unroll
      for (int mt = 0; mt < 4; ++mt)
        a[mt] = ldA(A, T_, mt * 16, kt * 32, lane);
#pragma unroll
      for (int nt = 0; nt < 4; ++nt)
        bb[nt] = ldB(Bt, T_, nt * 16, kt * 32, lane);
#pragma unroll
      for (int mt = 0; mt < 4; ++mt)
#pragma unroll
        for (int nt = 0; nt < 4; ++nt)
          WMMA_F16(acc[mt][nt], a[mt], bb[nt]);
    }
#pragma unroll
    for (int mt = 0; mt < 4; ++mt)
#pragma unroll
      for (int nt = 0; nt < 4; ++nt) {
        int cc = h * DH_ + nt * 16 + (lane & 15);
        int mo = mt * 16 + ((lane & 16) ? 8 : 0);
#pragma unroll
        for (int i = 0; i < 8; ++i) {
          float u = acc[mt][nt][i];
          float e = u > 0.f ? u : (__expf(u) - 1.f);   // ELU
          float v = copysignf(sqrtf(fabsf(e)), e);     // signed sqrt
          sTmp[(mo + i) * C_ + cc] = (_Float16)v;
        }
      }
  }
  __syncthreads();

  // ---- Stage 6: L2 normalize over T (per column), scale sTmp in place -----
  for (int c = tid; c < C_; c += 256) {
    float s = 0.f;
    for (int t = 0; t < T_; ++t) {
      float v = (float)sTmp[t * C_ + c];
      s += v * v;
    }
    rsc[c] = 1.f / fmaxf(sqrtf(s), 1e-12f);
  }
  __syncthreads();
  {
    int r = tid >> 2, q = tid & 3;
#pragma unroll 4
    for (int i = 0; i < 128; ++i) {
      int c = q * 128 + i;
      sTmp[r * C_ + c] = (_Float16)((float)sTmp[r * C_ + c] * rsc[c]);
    }
  }
  __syncthreads();

  // ---- Stage 7: GEMM3  y = xr + tmp @ W2 + b2  -> sY (f32) ----------------
  {
    const int col0 = wave * 64;
    v8f acc[4][4] = {};
    for (int kt = 0; kt < 16; ++kt) {
      if (kt + 2 < 16) {
        __builtin_prefetch(W2T + (size_t)(col0 + (lane & 15)) * C_ +
                               (kt + 2) * 32, 0, 0);
      }
      v16h a[4], bb[4];
#pragma unroll
      for (int mt = 0; mt < 4; ++mt)
        a[mt] = ldA(sTmp, C_, mt * 16, kt * 32, lane);
#pragma unroll
      for (int nt = 0; nt < 4; ++nt)
        bb[nt] = ldB(W2T, C_, col0 + nt * 16, kt * 32, lane);
#pragma unroll
      for (int mt = 0; mt < 4; ++mt)
#pragma unroll
        for (int nt = 0; nt < 4; ++nt)
          WMMA_F16(acc[mt][nt], a[mt], bb[nt]);
    }
#pragma unroll
    for (int mt = 0; mt < 4; ++mt)
#pragma unroll
      for (int nt = 0; nt < 4; ++nt) {
        int o = col0 + nt * 16 + (lane & 15);
        int mo = mt * 16 + ((lane & 16) ? 8 : 0);
        float bias = b2[o];
        const float* xp = x + ((size_t)(b * T_) * C_ + o) * N_ + n;
#pragma unroll
        for (int i = 0; i < 8; ++i) {
          int t = mo + i;
          float resid = xp[(size_t)t * C_ * N_];  // x[b,t,o,n] (L2 resident)
          sY[t * C_ + o] = acc[mt][nt][i] + bias + resid;
        }
      }
  }
  __syncthreads();

  // ---- Stage 8: LayerNorm(512) + scatter store ----------------------------
  {
    int r = tid >> 2, q = tid & 3;
    float s = 0.f, s2 = 0.f;
#pragma unroll 4
    for (int i = 0; i < 128; ++i) {
      float v = sY[r * C_ + q * 128 + i];
      s += v; s2 += v * v;
    }
    rowP[r * 4 + q] = s; rowP2[r * 4 + q] = s2;
  }
  __syncthreads();
  if (tid < T_) {
    float s  = rowP[tid * 4] + rowP[tid * 4 + 1] + rowP[tid * 4 + 2] + rowP[tid * 4 + 3];
    float s2 = rowP2[tid * 4] + rowP2[tid * 4 + 1] + rowP2[tid * 4 + 2] + rowP2[tid * 4 + 3];
    float mu = s * (1.f / (float)C_);
    float var = s2 * (1.f / (float)C_) - mu * mu;
    rowMu[tid] = mu;
    rowRs[tid] = rsqrtf(var + 1e-5f);
  }
  __syncthreads();
  {
    int r = tid >> 2, q = tid & 3;
    float mu = rowMu[r], rstd = rowRs[r];
    float* op = out + ((size_t)(b * T_ + r) * C_) * N_ + n;
#pragma unroll 4
    for (int i = 0; i < 128; ++i) {
      int c = q * 128 + i;
      float v = (sY[r * C_ + c] - mu) * rstd * ln_g[c] + ln_b[c];
      op[(size_t)c * N_] = v;
    }
  }
}

extern "C" void kernel_launch(void* const* d_in, const int* in_sizes, int n_in,
                              void* d_out, int out_size, void* d_ws, size_t ws_size,
                              hipStream_t stream) {
  const float* x   = (const float*)d_in[0];
  const float* Wg  = (const float*)d_in[1];
  const float* a1  = (const float*)d_in[2];
  const float* a2  = (const float*)d_in[3];
  const float* W2  = (const float*)d_in[4];
  const float* b2  = (const float*)d_in[5];
  const float* lg  = (const float*)d_in[6];
  const float* lb  = (const float*)d_in[7];
  float* out = (float*)d_out;

  _Float16* WgT = (_Float16*)d_ws;           // 512KB
  _Float16* W2T = WgT + (size_t)C_ * C_;     // 512KB

  fmt_prep<<<(C_ * C_ + 255) / 256, 256, 0, stream>>>(Wg, W2, WgT, W2T);

  const size_t smemBytes = 262144 + 10240;   // tiles + scratch (~266KB / 320KB WGP)
  hipFuncSetAttribute(reinterpret_cast<const void*>(fmt_main),
                      hipFuncAttributeMaxDynamicSharedMemorySize,
                      (int)smemBytes);
  fmt_main<<<4 * 14 * 14, 256, smemBytes, stream>>>(x, WgT, a1, a2, W2T, b2,
                                                    lg, lb, out);
}
